// MultiHeadAttention_27642409517143
// MI455X (gfx1250) — matrix operand order repaired and checked by reading, not compiled, not run on hardware
//
#include <hip/hip_runtime.h>
#include <hip/hip_bf16.h>

typedef __attribute__((ext_vector_type(16))) _Float16 v16h;
typedef __attribute__((ext_vector_type(8)))  float    v8f;

#define D_MODEL 512
#define NHEADS  8
#define DK      64
#define BATCH   8192
#define TB      16      // batch rows per workgroup

// ---------------------------------------------------------------------------
// B-fragment loader: B[k,n] = w[n*64 + k]  (w is [64][64] row-major f32,
// einsum "...k,jk->...j" means we multiply by w^T, i.e. B[k,n]=w[n,k]).
// 16-bit B 32x16 layout: lane l (ml=l&15, hi=l>>4), element e:
//   k = c*32 + hi*8 + (e<8 ? e : e+8) , n = t*16 + ml
// => two runs of 8 contiguous f32 per lane (32B aligned).
// ---------------------------------------------------------------------------
__device__ __forceinline__ v16h load_w_frag(const float* __restrict__ w,
                                            int c, int t, int ml, int hi) {
  const float* p = w + (t * 16 + ml) * DK + c * 32 + hi * 8;
  v16h r;
#pragma unroll
  for (int e = 0; e < 16; ++e) r[e] = (_Float16)p[e < 8 ? e : e + 8];
  return r;
}

__device__ __forceinline__ void load_head_weights(
    const float* __restrict__ w1, const float* __restrict__ b1,
    const float* __restrict__ w2, const float* __restrict__ b2,
    int ml, int hi, v16h (&bw)[2][4], float (&b1v)[4], float (&w2v)[4],
    float& b2v) {
#pragma unroll
  for (int c = 0; c < 2; ++c)
#pragma unroll
    for (int t = 0; t < 4; ++t) bw[c][t] = load_w_frag(w1, c, t, ml, hi);
#pragma unroll
  for (int t = 0; t < 4; ++t) {
    b1v[t] = b1[t * 16 + ml];
    w2v[t] = w2[t * 16 + ml];
  }
  b2v = b2[0];
}

// Pack per-lane f32 X (A-layout runs) into two 16x32 f16 A fragments.
// 16-bit A 16x32 layout: lane holds M=ml; element e of chunk c:
//   K = c*32 + hi*8 + (e<8 ? e : e+8)
// x[ri][e] holds K = hi*8 + (ri&1)*16 + (ri>>1)*32 + e.
__device__ __forceinline__ void pack_frag(const float (&x)[4][8], v16h (&a)[2]) {
#pragma unroll
  for (int c = 0; c < 2; ++c)
#pragma unroll
    for (int e = 0; e < 8; ++e) {
      a[c][e]     = (_Float16)x[2 * c][e];
      a[c][8 + e] = (_Float16)x[2 * c + 1][e];
    }
}

// D = X @ w1^T (16x64), sigmoid(+b1), dot w2, +b2 -> per-row score.
// D-layout: vgpr r, lane -> M = r + 8*hi, N = t*16 + ml.
// Reduce over N within each 16-lane half, broadcast via per-wave LDS row,
// return score for row M = ml (matching the A/att register layout).
__device__ __forceinline__ float pair_score(const v16h (&a)[2],
                                            const v16h (&bw)[2][4],
                                            const float (&b1v)[4],
                                            const float (&w2v)[4], float b2v,
                                            int ml, int hi,
                                            float* __restrict__ scrow) {
  v8f acc[4] = {};
#pragma unroll
  for (int c = 0; c < 2; ++c)
#pragma unroll
    for (int t = 0; t < 4; ++t)
      acc[t] = __builtin_amdgcn_wmma_f32_16x16x32_f16(
          false, a[c], false, bw[c][t], (short)0, acc[t], false, false);

  float s8[8];
#pragma unroll
  for (int r = 0; r < 8; ++r) s8[r] = 0.f;
#pragma unroll
  for (int t = 0; t < 4; ++t) {
#pragma unroll
    for (int r = 0; r < 8; ++r) {
      float v  = acc[t][r] + b1v[t];
      float sg = __builtin_amdgcn_rcpf(1.f + __expf(-v));  // sigmoid
      s8[r] += sg * w2v[t];
    }
  }
  // reduce over the 16 lanes of this half (rows are per-half in D layout)
#pragma unroll
  for (int r = 0; r < 8; ++r) {
#pragma unroll
    for (int m = 1; m < 16; m <<= 1) s8[r] += __shfl_xor(s8[r], m, 32);
  }
  if (ml == 0) {  // lanes 0 and 16 publish rows 0..7 / 8..15
#pragma unroll
    for (int r = 0; r < 8; ++r) scrow[hi * 8 + r] = s8[r] + b2v;
  }
  return scrow[ml];
}

// ---------------------------------------------------------------------------
// Kernel 1: per-pair score GEMMs + attention accumulation + zl/zs gating.
// Writes gated Y[8192][512] as f16 into workspace.
// ---------------------------------------------------------------------------
__global__ __launch_bounds__(256) void fuse_att_kernel(
    const float* __restrict__ user, const float* __restrict__ mood,
    const float* __restrict__ genre,
    const float* __restrict__ l_w1, const float* __restrict__ l_b1,
    const float* __restrict__ l_w2, const float* __restrict__ l_b2,
    const float* __restrict__ s_w1, const float* __restrict__ s_b1,
    const float* __restrict__ s_w2, const float* __restrict__ s_b2,
    const float* __restrict__ zl_w1, const float* __restrict__ zl_b1,
    const float* __restrict__ zl_w2, const float* __restrict__ zl_b2,
    const float* __restrict__ zs_w1, const float* __restrict__ zs_b1,
    const float* __restrict__ zs_w2, const float* __restrict__ zs_b2,
    _Float16* __restrict__ Y) {
  __shared__ float scbuf[NHEADS][16];

  const int tid  = threadIdx.x;
  const int wv   = tid >> 5;        // wave index == head
  const int lane = tid & 31;
  const int ml   = lane & 15;
  const int hi   = lane >> 4;
  const int b0   = blockIdx.x * TB;
  const int rowb = b0 + ml;         // this lane's batch row (M = ml)
  const int cb   = wv * DK;         // head's column base in D_MODEL
  const float scale = 0.125f;       // 1/sqrt(64)

  float* scrow = scbuf[wv];

  v16h  bw[2][4];
  float b1v[4], w2v[4], b2v;
  float attL[4][8] = {};
  float attS[4][8] = {};

  // ---------------- phase L: 4x5 (user,genre) pairs --------------------
  load_head_weights(l_w1, l_b1, l_w2, l_b2, ml, hi, bw, b1v, w2v, b2v);
  for (int u = 0; u < 4; ++u) {
    const float* urow = user + (size_t)(u * BATCH + rowb) * D_MODEL + cb;
    for (int g = 0; g < 5; ++g) {
      const float* grow = genre + (size_t)(g * BATCH + rowb) * D_MODEL + cb;
      float x[4][8];
#pragma unroll
      for (int ri = 0; ri < 4; ++ri) {
        const int ks = hi * 8 + (ri & 1) * 16 + (ri >> 1) * 32;
#pragma unroll
        for (int e = 0; e < 8; ++e) x[ri][e] = scale * urow[ks + e] * grow[ks + e];
      }
      v16h a[2];
      pack_frag(x, a);
      const float sc = pair_score(a, bw, b1v, w2v, b2v, ml, hi, scrow);
#pragma unroll
      for (int ri = 0; ri < 4; ++ri)
#pragma unroll
        for (int e = 0; e < 8; ++e) attL[ri][e] += sc * x[ri][e];
    }
  }

  // ---------------- phase S: 3x5 (mood,genre) pairs --------------------
  load_head_weights(s_w1, s_b1, s_w2, s_b2, ml, hi, bw, b1v, w2v, b2v);
  for (int m = 0; m < 3; ++m) {
    const float* mrow = mood + (size_t)(m * BATCH + rowb) * D_MODEL + cb;
    for (int g = 0; g < 5; ++g) {
      const float* grow = genre + (size_t)(g * BATCH + rowb) * D_MODEL + cb;
      float x[4][8];
#pragma unroll
      for (int ri = 0; ri < 4; ++ri) {
        const int ks = hi * 8 + (ri & 1) * 16 + (ri >> 1) * 32;
#pragma unroll
        for (int e = 0; e < 8; ++e) x[ri][e] = scale * mrow[ks + e] * grow[ks + e];
      }
      v16h a[2];
      pack_frag(x, a);
      const float sc = pair_score(a, bw, b1v, w2v, b2v, ml, hi, scrow);
#pragma unroll
      for (int ri = 0; ri < 4; ++ri)
#pragma unroll
        for (int e = 0; e < 8; ++e) attS[ri][e] += sc * x[ri][e];
    }
  }

  // ---------------- zl / zs scores + softmax gate ----------------------
  v16h aL[2], aS[2];
  pack_frag(attL, aL);
  pack_frag(attS, aS);

  load_head_weights(zl_w1, zl_b1, zl_w2, zl_b2, ml, hi, bw, b1v, w2v, b2v);
  const float rl = pair_score(aL, bw, b1v, w2v, b2v, ml, hi, scrow);

  load_head_weights(zs_w1, zs_b1, zs_w2, zs_b2, ml, hi, bw, b1v, w2v, b2v);
  const float rs = pair_score(aS, bw, b1v, w2v, b2v, ml, hi, scrow);

  const float mx  = fmaxf(rl, rs);
  const float e0  = __expf(rl - mx);
  const float e1  = __expf(rs - mx);
  const float inv = __builtin_amdgcn_rcpf(e0 + e1);
  const float g0  = e0 * inv;
  const float g1  = e1 * inv;

  // gated output Y[rowb, cb + k] (f16, same per-lane K runs as A layout)
#pragma unroll
  for (int ri = 0; ri < 4; ++ri) {
    const int ks = hi * 8 + (ri & 1) * 16 + (ri >> 1) * 32;
    _Float16* yp = Y + (size_t)rowb * D_MODEL + cb + ks;
#pragma unroll
    for (int e = 0; e < 8; ++e)
      yp[e] = (_Float16)(g0 * attL[ri][e] + g1 * attS[ri][e]);
  }
}

// ---------------------------------------------------------------------------
// Kernel 2: out = Y @ lin_w^T + lin_b   ([8192,512] x [512,512]^T)
// Block: 16 rows of Y staged in LDS (16 KB) shared by 8 waves; wave w owns
// output columns [w*64, w*64+64). f16 WMMA, f32 accumulate.
// ---------------------------------------------------------------------------
__global__ __launch_bounds__(256) void out_linear_kernel(
    const _Float16* __restrict__ Y, const float* __restrict__ lin_w,
    const float* __restrict__ lin_b, float* __restrict__ out) {
  __shared__ _Float16 ytile[TB * D_MODEL];

  const int tid  = threadIdx.x;
  const int wv   = tid >> 5;
  const int lane = tid & 31;
  const int ml   = lane & 15;
  const int hi   = lane >> 4;
  const int b0   = blockIdx.x * TB;

  // cooperative load of the 16x512 f16 Y tile (1024 x 16B)
  {
    const uint4* src = (const uint4*)(Y + (size_t)b0 * D_MODEL);
    uint4*       dst = (uint4*)ytile;
#pragma unroll
    for (int i = 0; i < 4; ++i) dst[tid + i * 256] = src[tid + i * 256];
  }
  __syncthreads();

  v8f acc[4] = {};
  for (int kk = 0; kk < D_MODEL; kk += 32) {
    // A fragment from LDS: A[m,k] = ytile[ml*512 + k]
    v16h a;
    const _Float16* yp = ytile + ml * D_MODEL + kk + hi * 8;
#pragma unroll
    for (int e = 0; e < 8; ++e) {
      a[e]     = yp[e];
      a[8 + e] = yp[16 + e];
    }
    // B fragments: B[k,n] = lin_w[n*512 + k], n = wv*64 + t*16 + ml
#pragma unroll
    for (int t = 0; t < 4; ++t) {
      const float* wp =
          lin_w + (size_t)(wv * DK + t * 16 + ml) * D_MODEL + kk + hi * 8;
      v16h bf;
#pragma unroll
      for (int e = 0; e < 16; ++e) bf[e] = (_Float16)wp[e < 8 ? e : e + 8];
      acc[t] = __builtin_amdgcn_wmma_f32_16x16x32_f16(
          false, a, false, bf, (short)0, acc[t], false, false);
    }
  }

  // D layout: vgpr r, lane -> row b0 + r + 8*hi, col n = wv*64 + t*16 + ml
#pragma unroll
  for (int t = 0; t < 4; ++t) {
    const int   n    = wv * DK + t * 16 + ml;
    const float bias = lin_b[n];
#pragma unroll
    for (int r = 0; r < 8; ++r)
      out[(size_t)(b0 + r + 8 * hi) * D_MODEL + n] = acc[t][r] + bias;
  }
}

// ---------------------------------------------------------------------------
extern "C" void kernel_launch(void* const* d_in, const int* in_sizes, int n_in,
                              void* d_out, int out_size, void* d_ws,
                              size_t ws_size, hipStream_t stream) {
  const float* user  = (const float*)d_in[0];
  const float* mood  = (const float*)d_in[1];
  const float* genre = (const float*)d_in[2];
  const float* l_w1  = (const float*)d_in[3];
  const float* l_b1  = (const float*)d_in[4];
  const float* l_w2  = (const float*)d_in[5];
  const float* l_b2  = (const float*)d_in[6];
  const float* s_w1  = (const float*)d_in[7];
  const float* s_b1  = (const float*)d_in[8];
  const float* s_w2  = (const float*)d_in[9];
  const float* s_b2  = (const float*)d_in[10];
  const float* zl_w1 = (const float*)d_in[11];
  const float* zl_b1 = (const float*)d_in[12];
  const float* zl_w2 = (const float*)d_in[13];
  const float* zl_b2 = (const float*)d_in[14];
  const float* zs_w1 = (const float*)d_in[15];
  const float* zs_b1 = (const float*)d_in[16];
  const float* zs_w2 = (const float*)d_in[17];
  const float* zs_b2 = (const float*)d_in[18];
  const float* lin_w = (const float*)d_in[19];
  const float* lin_b = (const float*)d_in[20];

  _Float16* Yws = (_Float16*)d_ws;  // 8192*512 f16 = 8 MB
  float*    out = (float*)d_out;

  fuse_att_kernel<<<BATCH / TB, 256, 0, stream>>>(
      user, mood, genre, l_w1, l_b1, l_w2, l_b2, s_w1, s_b1, s_w2, s_b2,
      zl_w1, zl_b1, zl_w2, zl_b2, zs_w1, zs_b1, zs_w2, zs_b2, Yws);

  out_linear_kernel<<<BATCH / TB, 256, 0, stream>>>(Yws, lin_w, lin_b, out);
}
